// BitNetQDyTAttention_66254165508543
// MI455X (gfx1250) — matrix operand order repaired and checked by reading, not compiled, not run on hardware
//
#include <hip/hip_runtime.h>
#include <hip/hip_bf16.h>
#include <stdint.h>

// ---------------- problem constants ----------------
#define DMODEL   768
#define NHEAD    12
#define HEADDIM  64
#define BATCH    16
#define SEQ      512
#define MROWS    (BATCH * SEQ)      // 8192
#define L4Q      7.0f
#define PCTQ     99.7f

// ---------------- vector types ----------------
typedef _Float16 v16h __attribute__((ext_vector_type(16)));
typedef _Float16 v8h  __attribute__((ext_vector_type(8)));
typedef float    v8f  __attribute__((ext_vector_type(8)));
typedef unsigned int u32x4 __attribute__((ext_vector_type(4)));
typedef int          i32x4 __attribute__((ext_vector_type(4)));
typedef int          i32x8 __attribute__((ext_vector_type(8)));

#if defined(__has_builtin)
#  if __has_builtin(__builtin_amdgcn_tensor_load_to_lds) && __has_builtin(__builtin_amdgcn_s_wait_tensorcnt)
#    define HAVE_TDM 1
#  else
#    define HAVE_TDM 0
#  endif
#else
#  define HAVE_TDM 0
#endif

// WMMA wrapper: D = A(16x32 f16) * B(32x16 f16) + C(16x16 f32)
__device__ __forceinline__ v8f wmma_f16(v16h a, v16h b, v8f c) {
    return __builtin_amdgcn_wmma_f32_16x16x32_f16(
        /*neg_a=*/false, a, /*neg_b=*/false, b,
        /*c_mod=*/(short)0, c, /*reuse_a=*/false, /*reuse_b=*/false);
}

// Load 16 halves as two 16-byte chunks separated by `gap` halves.
// gap=16 -> AMD table k order (K blocks {0..7},{16..23} per half-wave); unused, A uses gap=8
// gap=8  -> 16 consecutive halves (B-fragment, contiguous K)
__device__ __forceinline__ v16h ld16_split(const _Float16* p, int gap) {
    v8h a = *(const v8h*)p;
    v8h b = *(const v8h*)(p + gap);
    return __builtin_shufflevector(a, b, 0,1,2,3,4,5,6,7,8,9,10,11,12,13,14,15);
}

// Strided gather of 16 halves (B-fragment when K runs along rows)
__device__ __forceinline__ v16h ld16_strided(const _Float16* p, int ld) {
    v16h r;
#pragma unroll
    for (int j = 0; j < 16; ++j) r[j] = p[(size_t)j * ld];
    return r;
}

// =====================================================================
// TDM staging: DMA a 64-row x 64-half tile (row stride DMODEL halves)
// from global memory into LDS. 2-D D# per ISA 8.3/8.4.
// =====================================================================
__device__ __forceinline__ void tdm_stage_tile(const _Float16* gsrc,
                                               _Float16* lds_dst,
                                               int wave, int tid) {
#if HAVE_TDM
    if (wave == 0) {
        unsigned long long ga = (unsigned long long)(const void*)gsrc;
        // LDS aperture maps addr[31:0] directly to the LDS offset
        unsigned lds_off = (unsigned)(unsigned long long)(void*)lds_dst;

        u32x4 g0;
        g0[0] = 1u;                                   // count=1, user D#
        g0[1] = lds_off;                              // lds_addr (bytes)
        g0[2] = (unsigned)(ga & 0xFFFFFFFFull);       // global_addr[31:0]
        g0[3] = (unsigned)((ga >> 32) & 0x01FFFFFFull) | (2u << 30); // addr[56:32] | type=2

        i32x8 g1;
        g1[0] = (1 << 16);                            // data_size=1 (2 bytes/elem)
        g1[1] = (DMODEL & 0xFFFF) << 16;              // tensor_dim0 low16 = 768
        g1[2] = ((DMODEL >> 16) & 0xFFFF) |           // tensor_dim0 hi16
                ((MROWS & 0xFFFF) << 16);             // tensor_dim1 low16 = 8192
        g1[3] = ((MROWS >> 16) & 0xFFFF) |            // tensor_dim1 hi16
                (64 << 16);                           // tile_dim0 = 64
        g1[4] = 64;                                   // tile_dim1=64, tile_dim2=0
        g1[5] = DMODEL;                               // tensor_dim0_stride low32 = 768
        g1[6] = 0;                                    // stride hi / dim1_stride low
        g1[7] = 0;

        i32x4 g2; g2[0] = 0; g2[1] = 0; g2[2] = 0; g2[3] = 0;
        i32x4 g3; g3[0] = 0; g3[1] = 0; g3[2] = 0; g3[3] = 0;

#if defined(__clang_major__) && (__clang_major__ >= 23)
        i32x8 gz; gz[0]=0; gz[1]=0; gz[2]=0; gz[3]=0; gz[4]=0; gz[5]=0; gz[6]=0; gz[7]=0;
        __builtin_amdgcn_tensor_load_to_lds(g0, g1, g2, g3, gz, 0);
#else
        __builtin_amdgcn_tensor_load_to_lds(g0, g1, g2, g3, 0);
#endif
    }
#else
    // fallback: cooperative b128 staging by all 128 threads
    for (int i = tid; i < 64 * 8; i += 128) {      // 512 chunks of 8 halves
        int r = i >> 3, c8 = i & 7;
        *(v8h*)&lds_dst[r * 64 + c8 * 8] =
            *(const v8h*)(gsrc + (size_t)r * DMODEL + c8 * 8);
    }
#endif
}

__device__ __forceinline__ void tdm_wait_current(int wave, int more_pending) {
#if HAVE_TDM
    if (wave == 0) {
        if (more_pending) __builtin_amdgcn_s_wait_tensorcnt(2); // next 2 still in flight
        else              __builtin_amdgcn_s_wait_tensorcnt(0);
    }
#endif
}

// =====================================================================
// K1: block-64 FWHT + sign1/perm/sign2  (one workgroup per row)
// =====================================================================
__global__ void k_fwht_perm(const float* __restrict__ hs,
                            const float* __restrict__ sign1,
                            const float* __restrict__ sign2,
                            const int*   __restrict__ perm,
                            float* __restrict__ xt) {
    __shared__ float row[DMODEL];
    const int r   = blockIdx.x;
    const int tid = threadIdx.x;
    const size_t base = (size_t)r * DMODEL;

    for (int i = tid; i < DMODEL; i += 256) row[i] = hs[base + i];
    __syncthreads();

#pragma unroll
    for (int stride = 1; stride < 64; stride <<= 1) {
        for (int i = tid; i < DMODEL / 2; i += 256) {
            int blk = i >> 5;
            int j   = i & 31;
            int q   = j / stride;
            int s   = j % stride;
            int lo  = blk * 64 + q * 2 * stride + s;
            int hi  = lo + stride;
            float a = row[lo], b = row[hi];
            row[lo] = a + b;
            row[hi] = a - b;
        }
        __syncthreads();
    }

    for (int j = tid; j < DMODEL; j += 256) {
        int p = perm[j];
        xt[base + j] = row[p] * 0.125f * sign1[p] * sign2[j];  // 1/sqrt(64)
    }
}

// =====================================================================
// K2: exact per-channel linear-interpolated 99.7% quantile of |x|
// =====================================================================
__device__ uint32_t select_kth(const uint32_t* vals, uint32_t* red, int k, int tid) {
    uint32_t lo = 0u, hi = 0x7F800000u;
    while (lo < hi) {
        uint32_t mid = lo + ((hi - lo) >> 1);
        uint32_t cnt = 0;
        for (int i = tid; i < MROWS; i += 256) cnt += (vals[i] <= mid);
        red[tid] = cnt;
        __syncthreads();
        for (int s = 128; s > 0; s >>= 1) {
            if (tid < s) red[tid] += red[tid + s];
            __syncthreads();
        }
        uint32_t total = red[0];
        __syncthreads();
        if (total >= (uint32_t)(k + 1)) hi = mid; else lo = mid + 1;
    }
    return lo;
}

__global__ void k_quantile(const float* __restrict__ xt, float* __restrict__ scale) {
    __shared__ uint32_t vals[MROWS];
    __shared__ uint32_t red[256];
    const int c   = blockIdx.x;
    const int tid = threadIdx.x;

    for (int r = tid; r < MROWS; r += 256)
        vals[r] = __float_as_uint(fabsf(xt[(size_t)r * DMODEL + c]));
    __syncthreads();

    const float p  = (PCTQ / 100.0f) * (float)(MROWS - 1);
    const int   k0 = (int)floorf(p);
    const float fr = p - (float)k0;

    uint32_t b0 = select_kth(vals, red, k0,     tid);
    uint32_t b1 = select_kth(vals, red, k0 + 1, tid);

    if (tid == 0) {
        float q = __uint_as_float(b0) * (1.0f - fr) + __uint_as_float(b1) * fr;
        scale[c] = q / L4Q;
    }
}

// =====================================================================
// K3: int4 fake-quant activations -> f16
// =====================================================================
__global__ void k_quant_x(const float* __restrict__ xt,
                          const float* __restrict__ scale,
                          _Float16* __restrict__ xq) {
    size_t idx = (size_t)blockIdx.x * blockDim.x + threadIdx.x;
    if (idx >= (size_t)MROWS * DMODEL) return;
    int c = (int)(idx % DMODEL);
    float s  = scale[c];
    float xs = xt[idx] / (s + 1e-8f);
    float q  = fminf(fmaxf(rintf(xs), -L4Q), L4Q) * s;
    xq[idx] = (_Float16)q;
}

// =====================================================================
// K4: ternary weight quantization -> f16 (4 matrices, row-major)
// =====================================================================
__global__ void k_quant_w(const float* __restrict__ w,
                          const float* __restrict__ s_tilde,
                          const float* __restrict__ t,
                          const float* __restrict__ delta,
                          _Float16* __restrict__ wq) {
    size_t idx = (size_t)blockIdx.x * blockDim.x + threadIdx.x;
    if (idx >= (size_t)4 * DMODEL * DMODEL) return;
    int m = (int)(idx / (DMODEL * DMODEL));
    int o = (int)((idx / DMODEL) % DMODEL);
    float st = s_tilde[m * DMODEL + o];
    float s  = (st > 20.0f) ? st : log1pf(__expf(st));
    float diff = w[idx] - t[m * DMODEL + o];
    float sg = (diff > 0.0f) ? 1.0f : ((diff < 0.0f) ? -1.0f : 0.0f);
    float q  = sg * ((fabsf(diff) > delta[m * DMODEL + o]) ? 1.0f : 0.0f);
    wq[idx] = (_Float16)(s * q);
}

// =====================================================================
// K5/K7: WMMA GEMM  Y = Xq @ W[m]^T
// =====================================================================
__global__ void k_gemm_wmma(const _Float16* __restrict__ X,
                            const _Float16* __restrict__ Wbase,
                            _Float16* __restrict__ Yh,
                            float* __restrict__ Yf,
                            int out_is_f32) {
    const int m    = blockIdx.z;
    const _Float16* W = Wbase + (size_t)m * DMODEL * DMODEL;
    const int tid  = threadIdx.x;
    const int lane = tid & 31;
    const int wave = tid >> 5;
    const int wm   = wave & 3;
    const int wn   = wave >> 2;
    const int row0 = blockIdx.x * 64 + wm * 16;
    const int col0 = blockIdx.y * 64 + wn * 32;
    const int l    = lane & 15;
    const int hiw  = lane >> 4;

    v8f acc0 = {}, acc1 = {};

#pragma unroll
    for (int k0 = 0; k0 < DMODEL; k0 += 32) {
        v16h a  = ld16_split(X + (size_t)(row0 + l) * DMODEL + k0 + hiw * 16, 8);
        v16h b0 = ld16_split(W + (size_t)(col0 + l) * DMODEL + k0 + hiw * 16, 8);
        v16h b1 = ld16_split(W + (size_t)(col0 + 16 + l) * DMODEL + k0 + hiw * 16, 8);
        if (k0 + 64 < DMODEL) {
            __builtin_prefetch(W + (size_t)(col0 + l) * DMODEL + k0 + 64, 0, 1);
        }
        acc0 = wmma_f16(a, b0, acc0);
        acc1 = wmma_f16(a, b1, acc1);
    }

    if (out_is_f32) {
#pragma unroll
        for (int g = 0; g < 8; ++g) {
            int row = row0 + g + hiw * 8;
            Yf[(size_t)row * DMODEL + col0 + l]      = acc0[g];
            Yf[(size_t)row * DMODEL + col0 + 16 + l] = acc1[g];
        }
    } else {
        _Float16* y = Yh + (size_t)m * MROWS * DMODEL;
#pragma unroll
        for (int g = 0; g < 8; ++g) {
            int row = row0 + g + hiw * 8;
            y[(size_t)row * DMODEL + col0 + l]      = (_Float16)acc0[g];
            y[(size_t)row * DMODEL + col0 + 16 + l] = (_Float16)acc1[g];
        }
    }
}

// =====================================================================
// K6: flash-style attention with TDM double-buffered K/V staging.
//   One WG per (b,h,qblock64); 4 waves x 16 query rows.
//   LDS: K[2][64][64] + V[2][64][64] + P[4][16][64] = 40 KB.
// =====================================================================
__global__ void k_attention(const _Float16* __restrict__ QKV,
                            const float* __restrict__ mask,
                            _Float16* __restrict__ CTX) {
    const int bh   = blockIdx.x;
    const int b    = bh / NHEAD;
    const int h    = bh % NHEAD;
    const int qblk = blockIdx.y;
    const int tid  = threadIdx.x;
    const int lane = tid & 31;
    const int wave = tid >> 5;          // 0..3
    const int l    = lane & 15;
    const int hiw  = lane >> 4;

    const _Float16* Qg = QKV;
    const _Float16* Kg = QKV + (size_t)1 * MROWS * DMODEL;
    const _Float16* Vg = QKV + (size_t)2 * MROWS * DMODEL;

    const int qr0  = b * SEQ + qblk * 64 + wave * 16;
    const int hoff = h * HEADDIM;

    __shared__ _Float16 smem[2 * 4096 + 2 * 4096 + 4 * 1024]; // 40 KB
    _Float16* Kt0 = smem;
    _Float16* Kt1 = smem + 4096;
    _Float16* Vt0 = smem + 8192;
    _Float16* Vt1 = smem + 8192 + 4096;
    _Float16* Pw  = smem + 16384 + wave * 1024;

    // Q fragments (A layout), reused across all K blocks
    v16h aq0 = ld16_split(Qg + (size_t)(qr0 + l) * DMODEL + hoff + 0  + hiw * 16, 8);
    v16h aq1 = ld16_split(Qg + (size_t)(qr0 + l) * DMODEL + hoff + 32 + hiw * 16, 8);

    float mprev[8], lacc[8];
    v8f o[4] = {};
#pragma unroll
    for (int g = 0; g < 8; ++g) { mprev[g] = -1e30f; lacc[g] = 0.0f; }

    // prime buffer 0 with block 0
    {
        const int kr0 = b * SEQ;
        tdm_stage_tile(Kg + (size_t)kr0 * DMODEL + hoff, Kt0, wave, tid);
        tdm_stage_tile(Vg + (size_t)kr0 * DMODEL + hoff, Vt0, wave, tid);
    }

    for (int kb = 0; kb < SEQ / 64; ++kb) {
        __syncthreads();   // buffer (kb+1)&1 free of readers from iter kb-1

        const int more = (kb + 1 < SEQ / 64);
        if (more) {
            const int krn = b * SEQ + (kb + 1) * 64;
            _Float16* kdst = ((kb + 1) & 1) ? Kt1 : Kt0;
            _Float16* vdst = ((kb + 1) & 1) ? Vt1 : Vt0;
            tdm_stage_tile(Kg + (size_t)krn * DMODEL + hoff, kdst, wave, tid);
            tdm_stage_tile(Vg + (size_t)krn * DMODEL + hoff, vdst, wave, tid);
        }
        tdm_wait_current(wave, more);   // current buffer's 2 tiles retired
        __syncthreads();

        const _Float16* Kc = (kb & 1) ? Kt1 : Kt0;
        const _Float16* Vc = (kb & 1) ? Vt1 : Vt0;

        // ---- scores S = Q K^T : 4 tiles of 16x16, K from LDS ----
        v8f s[4] = {};
#pragma unroll
        for (int nt = 0; nt < 4; ++nt) {
            v16h b0 = ld16_split(Kc + (nt * 16 + l) * 64 + 0  + hiw * 16, 8);
            v16h b1 = ld16_split(Kc + (nt * 16 + l) * 64 + 32 + hiw * 16, 8);
            s[nt] = wmma_f16(aq1, b1, wmma_f16(aq0, b0, s[nt]));
        }

        // ---- scale + mask + row max ----
        float rm[8];
#pragma unroll
        for (int g = 0; g < 8; ++g) rm[g] = -1e30f;
#pragma unroll
        for (int nt = 0; nt < 4; ++nt) {
            float mv = mask[b * SEQ + kb * 64 + nt * 16 + l];
#pragma unroll
            for (int g = 0; g < 8; ++g) {
                float v = s[nt][g] * 0.125f + mv;   // 1/sqrt(64)
                s[nt][g] = v;
                rm[g] = fmaxf(rm[g], v);
            }
        }
#pragma unroll
        for (int msk = 1; msk < 16; msk <<= 1)
#pragma unroll
            for (int g = 0; g < 8; ++g)
                rm[g] = fmaxf(rm[g], __shfl_xor(rm[g], msk, 32));

        // ---- online softmax ----
        float corr[8], rs[8];
#pragma unroll
        for (int g = 0; g < 8; ++g) {
            float mn = fmaxf(mprev[g], rm[g]);
            corr[g]  = __expf(mprev[g] - mn);
            mprev[g] = mn;
            rs[g]    = 0.0f;
        }
#pragma unroll
        for (int nt = 0; nt < 4; ++nt) {
#pragma unroll
            for (int g = 0; g < 8; ++g) {
                float p = __expf(s[nt][g] - mprev[g]);
                rs[g] += p;
                Pw[(g + hiw * 8) * 64 + nt * 16 + l] = (_Float16)p;
            }
        }
#pragma unroll
        for (int msk = 1; msk < 16; msk <<= 1)
#pragma unroll
            for (int g = 0; g < 8; ++g)
                rs[g] += __shfl_xor(rs[g], msk, 32);
#pragma unroll
        for (int g = 0; g < 8; ++g) lacc[g] = lacc[g] * corr[g] + rs[g];

        // ---- O = corr*O + P @ V (P via LDS layout swap, V from LDS) ----
        v16h ap0 = ld16_split(Pw + l * 64 + 0  + hiw * 16, 8);
        v16h ap1 = ld16_split(Pw + l * 64 + 32 + hiw * 16, 8);
#pragma unroll
        for (int ot = 0; ot < 4; ++ot) {
#pragma unroll
            for (int g = 0; g < 8; ++g) o[ot][g] *= corr[g];
            v16h bv0 = ld16_strided(Vc + (0  + hiw * 16) * 64 + ot * 16 + l, 64);
            v16h bv1 = ld16_strided(Vc + (32 + hiw * 16) * 64 + ot * 16 + l, 64);
            o[ot] = wmma_f16(ap1, bv1, wmma_f16(ap0, bv0, o[ot]));
        }
    }

    // ---- normalize + write ctx (f16) ----
#pragma unroll
    for (int ot = 0; ot < 4; ++ot)
#pragma unroll
        for (int g = 0; g < 8; ++g) {
            int row = qr0 + g + hiw * 8;
            CTX[(size_t)row * DMODEL + hoff + ot * 16 + l] =
                (_Float16)(o[ot][g] / lacc[g]);
        }
}

// =====================================================================
// host-side launch
// =====================================================================
extern "C" void kernel_launch(void* const* d_in, const int* in_sizes, int n_in,
                              void* d_out, int out_size, void* d_ws, size_t ws_size,
                              hipStream_t stream) {
    const float* hs    = (const float*)d_in[0];
    const float* amask = (const float*)d_in[1];
    const float* w     = (const float*)d_in[2];
    const float* s_t   = (const float*)d_in[3];
    const float* t     = (const float*)d_in[4];
    const float* dlt   = (const float*)d_in[5];
    const float* sg1   = (const float*)d_in[6];
    const float* sg2   = (const float*)d_in[7];
    const int*   perm  = (const int*)d_in[8];
    float* out = (float*)d_out;

    char* ws = (char*)d_ws;
    const size_t OFF_XT    = 0;
    const size_t OFF_SCALE = OFF_XT    + (size_t)MROWS * DMODEL * 4;
    const size_t OFF_XQ    = OFF_SCALE + 4096;
    const size_t OFF_WH    = OFF_XQ    + (size_t)MROWS * DMODEL * 2;
    const size_t OFF_QKV   = OFF_WH    + (size_t)4 * DMODEL * DMODEL * 2;
    const size_t OFF_CTX   = OFF_QKV   + (size_t)3 * MROWS * DMODEL * 2;
    const size_t NEED      = OFF_CTX   + (size_t)MROWS * DMODEL * 2;
    if (ws_size < NEED) return;

    float*    xt    = (float*)(ws + OFF_XT);
    float*    scale = (float*)(ws + OFF_SCALE);
    _Float16* xq    = (_Float16*)(ws + OFF_XQ);
    _Float16* wh    = (_Float16*)(ws + OFF_WH);
    _Float16* qkv   = (_Float16*)(ws + OFF_QKV);
    _Float16* ctx   = (_Float16*)(ws + OFF_CTX);

    k_fwht_perm<<<MROWS, 256, 0, stream>>>(hs, sg1, sg2, perm, xt);
    k_quantile<<<DMODEL, 256, 0, stream>>>(xt, scale);
    {
        size_t n = (size_t)MROWS * DMODEL;
        k_quant_x<<<(unsigned)((n + 255) / 256), 256, 0, stream>>>(xt, scale, xq);
    }
    {
        size_t n = (size_t)4 * DMODEL * DMODEL;
        k_quant_w<<<(unsigned)((n + 255) / 256), 256, 0, stream>>>(w, s_t, t, dlt, wh);
    }
    {
        dim3 grid(MROWS / 64, DMODEL / 64, 3);
        k_gemm_wmma<<<grid, 256, 0, stream>>>(xq, wh, qkv, nullptr, 0);
    }
    {
        dim3 grid(BATCH * NHEAD, SEQ / 64);
        k_attention<<<grid, 128, 0, stream>>>(qkv, amask, ctx);
    }
    {
        dim3 grid(MROWS / 64, DMODEL / 64, 1);
        k_gemm_wmma<<<grid, 256, 0, stream>>>(ctx, wh + (size_t)3 * DMODEL * DMODEL,
                                              nullptr, out, 1);
    }
}
